// DeepSeekMoE_87600152969590
// MI455X (gfx1250) — matrix operand order repaired and checked by reading, not compiled, not run on hardware
//
#include <hip/hip_runtime.h>
#include <hip/hip_bf16.h>

// ---------------------------------------------------------------------------
// DeepSeek-MoE forward for MI455X (gfx1250, wave32, WMMA).
// Dense-expert formulation: y = sum_e (cw[:,e] * silu(h Wg_e^T) * (h Wu_e^T)) Wd_e^T
// Heavy math: v_wmma_f32_16x16x32_bf16.
// Data movement: global_load_async_to_lds_b128 (ASYNCcnt) when available.
// ---------------------------------------------------------------------------

#define TOK   4096          // B*S
#define DIM   2048          // D
#define MD    1408          // MOE_D
#define NEXP  16            // E
#define NTOP  6             // K
#define BK    32            // WMMA K-step (bf16)

typedef __bf16 bf16;
typedef __attribute__((ext_vector_type(8)))  __bf16 v8bf;
typedef __attribute__((ext_vector_type(16))) __bf16 v16bf;
typedef __attribute__((ext_vector_type(8)))  float  v8f;
typedef __attribute__((ext_vector_type(4)))  int    v4i;

// ---- CDNA5 async memory->LDS path -----------------------------------------
#if defined(__has_builtin)
#  if __has_builtin(__builtin_amdgcn_global_load_async_to_lds_b128)
#    define USE_ASYNC_LDS 1
#  endif
#endif
#ifndef USE_ASYNC_LDS
#  define USE_ASYNC_LDS 0
#endif

#if USE_ASYNC_LDS
typedef __attribute__((address_space(1))) v4i GV4;   // global int4
typedef __attribute__((address_space(3))) v4i LV4;   // LDS int4
__device__ __forceinline__ void async_cp16(const void* g, void* l) {
    // generic->AS casts via integers (low 32 bits of a generic LDS pointer are
    // the LDS byte offset on AMDGPU).
    __builtin_amdgcn_global_load_async_to_lds_b128(
        (GV4*)(unsigned long long)g,
        (LV4*)(unsigned)(unsigned long long)l, 0, 0);
}
#  if __has_builtin(__builtin_amdgcn_s_wait_asynccnt)
#    define WAIT_ASYNC0() __builtin_amdgcn_s_wait_asynccnt(0)
#  else
#    define WAIT_ASYNC0() asm volatile("s_wait_asynccnt 0x0" ::: "memory")
#  endif
#endif

// ---- workspace layout (bytes) ---------------------------------------------
static constexpr size_t H_ELEMS   = (size_t)TOK * DIM;          // 8,388,608
static constexpr size_t W_ELEMS   = (size_t)NEXP * MD * DIM;    // 46,137,344
static constexpr size_t ACT_ELEMS = (size_t)NEXP * TOK * MD;    // 92,274,688
static constexpr size_t CW_ELEMS  = (size_t)TOK * NEXP;

static constexpr size_t H_OFF   = 0;
static constexpr size_t WG_OFF  = H_OFF  + 2 * H_ELEMS;
static constexpr size_t WU_OFF  = WG_OFF + 2 * W_ELEMS;
static constexpr size_t WD_OFF  = WU_OFF + 2 * W_ELEMS;
static constexpr size_t ACT_OFF = WD_OFF + 2 * W_ELEMS;
static constexpr size_t CW_OFF  = ACT_OFF + 2 * ACT_ELEMS;

// ---- helpers --------------------------------------------------------------
__device__ __forceinline__ unsigned short f2bf(float f) {
    unsigned u = __float_as_uint(f);
    u += 0x7FFFu + ((u >> 16) & 1u);          // round-to-nearest-even
    return (unsigned short)(u >> 16);
}

__device__ __forceinline__ v8f wmma_bf16(v16bf a, v16bf b, v8f c) {
    // (neg_a, A, neg_b, B, c_mod, C, reuse_a, reuse_b)
    return __builtin_amdgcn_wmma_f32_16x16x32_bf16(false, a, false, b,
                                                   (short)0, c, false, false);
}

// A fragment: LDS tile [M][32] row-major. Lane L<16 -> row M=L, K {0..7,16..23};
// lane L+16 -> row M=L, K {8..15,24..31}. Two 16-byte ds loads per lane.
__device__ __forceinline__ v16bf ldsA(const bf16* base, int m0, int lane) {
    const int r  = lane & 15;
    const int hi = lane >> 4;
    const bf16* p = base + (m0 + r) * BK + hi * 8;
    v8bf a = *(const v8bf*)p;
    v8bf b = *(const v8bf*)(p + 16);
    return __builtin_shufflevector(a, b, 0,1,2,3,4,5,6,7,8,9,10,11,12,13,14,15);
}

// B fragment: LDS tile [N][32] row-major ([out][in] weight layout).
// Lane L<16 -> col N=L, K {0..7,16..23} ; lane L+16 -> col N=L, K {8..15,24..31}.
__device__ __forceinline__ v16bf ldsB(const bf16* base, int n0, int lane) {
    const int c  = lane & 15;
    const int hi = lane >> 4;
    const bf16* p = base + (n0 + c) * BK + hi * 8;
    v8bf a = *(const v8bf*)p;
    v8bf b = *(const v8bf*)(p + 16);
    return __builtin_shufflevector(a, b, 0,1,2,3,4,5,6,7,8,9,10,11,12,13,14,15);
}

// ---- kernel 1: fp32 -> bf16 conversion ------------------------------------
__global__ __launch_bounds__(256) void cvt_f32_bf16(const float4* __restrict__ in,
                                                    uint2* __restrict__ out, int n4) {
    int i = blockIdx.x * 256 + threadIdx.x;
    if (i >= n4) return;
    float4 v = in[i];
    uint2 o;
    o.x = (unsigned)f2bf(v.x) | ((unsigned)f2bf(v.y) << 16);
    o.y = (unsigned)f2bf(v.z) | ((unsigned)f2bf(v.w) << 16);
    out[i] = o;
}

// ---- kernel 2: router (softmax + top-6 -> dense combine weights) ----------
__global__ __launch_bounds__(256) void router_kernel(const float* __restrict__ x,
                                                     const float* __restrict__ gw,
                                                     float* __restrict__ cw) {
    const int lane = threadIdx.x & 31;
    const int t    = blockIdx.x * 8 + (threadIdx.x >> 5);   // one wave per token

    float acc[NEXP];
#pragma unroll
    for (int e = 0; e < NEXP; ++e) acc[e] = 0.f;

    const float* xr = x + (size_t)t * DIM;
    for (int i = 0; i < DIM / 32; ++i) {
        const int k = lane + i * 32;
        const float hv = xr[k];
#pragma unroll
        for (int e = 0; e < NEXP; ++e) acc[e] += hv * gw[e * DIM + k];
    }
#pragma unroll
    for (int e = 0; e < NEXP; ++e) {
        acc[e] += __shfl_xor(acc[e], 16, 32);
        acc[e] += __shfl_xor(acc[e],  8, 32);
        acc[e] += __shfl_xor(acc[e],  4, 32);
        acc[e] += __shfl_xor(acc[e],  2, 32);
        acc[e] += __shfl_xor(acc[e],  1, 32);
    }
    if (lane == 0) {
        float mx = acc[0];
#pragma unroll
        for (int e = 1; e < NEXP; ++e) mx = fmaxf(mx, acc[e]);
        float sc[NEXP];
        float sum = 0.f;
#pragma unroll
        for (int e = 0; e < NEXP; ++e) { sc[e] = __expf(acc[e] - mx); sum += sc[e]; }
        const float inv = 1.f / sum;
#pragma unroll
        for (int e = 0; e < NEXP; ++e) sc[e] *= inv;
#pragma unroll
        for (int e = 0; e < NEXP; ++e) {
            int rank = 0;
#pragma unroll
            for (int j = 0; j < NEXP; ++j)
                rank += (sc[j] > sc[e]) || (sc[j] == sc[e] && j < e);
            cw[(size_t)t * NEXP + e] = (rank < NTOP) ? sc[e] : 0.f;
        }
    }
}

// ---- kernel 3: fused gate+up GEMM + SwiGLU + cw scaling -> act (bf16) -----
// Block tile 128(M) x 64(N), K-step 32, 8 waves (2x4), wave tile 64x16.
__global__ __launch_bounds__(256) void moe_gateup(const bf16* __restrict__ hB,
                                                  const bf16* __restrict__ wgB,
                                                  const bf16* __restrict__ wuB,
                                                  const float* __restrict__ cw,
                                                  unsigned short* __restrict__ act) {
    __shared__ __align__(16) bf16 As[2][128 * BK];
    __shared__ __align__(16) bf16 Bg[2][64 * BK];
    __shared__ __align__(16) bf16 Bu[2][64 * BK];
    __shared__ float cws[128];

    const int tid  = threadIdx.x;
    const int lane = tid & 31;
    const int wave = tid >> 5;
    const int wm   = wave & 1;         // 2 M-groups of 64
    const int wn   = wave >> 1;        // 4 N-groups of 16
    const int m0   = blockIdx.x * 128;
    const int n0   = blockIdx.y * 64;
    const int e    = blockIdx.z;

    const bf16* wgE = wgB + (size_t)e * MD * DIM;
    const bf16* wuE = wuB + (size_t)e * MD * DIM;

    if (tid < 128) cws[tid] = cw[(size_t)(m0 + tid) * NEXP + e];

    // per-thread 16B chunks: A = 512 chunks (2/thr), Bg/Bu = 256 each (1/thr)
    const int arow0 = tid >> 2, aoff = (tid & 3) * 8, arow1 = arow0 + 64;
    const int brow  = tid >> 2, boff = aoff;
    const int NKT   = DIM / BK;        // 64

#if USE_ASYNC_LDS
    auto issueTile = [&](int kt, int buf) {
        const bf16* asrc = hB + (size_t)m0 * DIM + kt * BK;
        async_cp16(asrc + (size_t)arow0 * DIM + aoff, As[buf] + arow0 * BK + aoff);
        async_cp16(asrc + (size_t)arow1 * DIM + aoff, As[buf] + arow1 * BK + aoff);
        async_cp16(wgE + (size_t)(n0 + brow) * DIM + kt * BK + boff, Bg[buf] + brow * BK + boff);
        async_cp16(wuE + (size_t)(n0 + brow) * DIM + kt * BK + boff, Bu[buf] + brow * BK + boff);
    };
#else
    uint4 ra0, ra1, rg, ru;
    auto loadRegs = [&](int kt) {
        const bf16* asrc = hB + (size_t)m0 * DIM + kt * BK;
        ra0 = *(const uint4*)(asrc + (size_t)arow0 * DIM + aoff);
        ra1 = *(const uint4*)(asrc + (size_t)arow1 * DIM + aoff);
        rg  = *(const uint4*)(wgE + (size_t)(n0 + brow) * DIM + kt * BK + boff);
        ru  = *(const uint4*)(wuE + (size_t)(n0 + brow) * DIM + kt * BK + boff);
    };
    auto storeLds = [&](int buf) {
        *(uint4*)(As[buf] + arow0 * BK + aoff) = ra0;
        *(uint4*)(As[buf] + arow1 * BK + aoff) = ra1;
        *(uint4*)(Bg[buf] + brow * BK + boff)  = rg;
        *(uint4*)(Bu[buf] + brow * BK + boff)  = ru;
    };
#endif

    const v8f vz = {0.f,0.f,0.f,0.f,0.f,0.f,0.f,0.f};
    v8f accG[4], accU[4];
#pragma unroll
    for (int mf = 0; mf < 4; ++mf) { accG[mf] = vz; accU[mf] = vz; }

    int buf = 0;
#if USE_ASYNC_LDS
    issueTile(0, 0);
    for (int kt = 0; kt < NKT; ++kt) {
        WAIT_ASYNC0();                       // my copies for tile kt landed in LDS
        __syncthreads();                     // whole block's tile kt resident;
                                             // everyone done reading buf^1
        if (kt + 1 < NKT) issueTile(kt + 1, buf ^ 1);   // overlaps compute below
#else
    loadRegs(0);
    for (int kt = 0; kt < NKT; ++kt) {
        storeLds(buf);
        if (kt + 1 < NKT) loadRegs(kt + 1);
        __syncthreads();
#endif
        v16bf fg = ldsB(Bg[buf], wn * 16, lane);
        v16bf fu = ldsB(Bu[buf], wn * 16, lane);
#pragma unroll
        for (int mf = 0; mf < 4; ++mf) {
            v16bf fa = ldsA(As[buf], wm * 64 + mf * 16, lane);
            accG[mf] = wmma_bf16(fa, fg, accG[mf]);
            accU[mf] = wmma_bf16(fa, fu, accU[mf]);
        }
        buf ^= 1;
    }

    // epilogue: act = cw * silu(g) * u  (C layout: VGPR r -> M = hi*8 + r)
    const int hi8 = (lane >> 4) * 8;
    const int col = n0 + wn * 16 + (lane & 15);
    unsigned short* actE = act + (size_t)e * TOK * MD;
#pragma unroll
    for (int mf = 0; mf < 4; ++mf) {
#pragma unroll
        for (int r = 0; r < 8; ++r) {
            const int rowb = wm * 64 + mf * 16 + hi8 + r;
            const float g = accG[mf][r], u = accU[mf][r];
            const float sv = g / (1.0f + __expf(-g));
            const float v  = cws[rowb] * sv * u;
            actE[(size_t)(m0 + rowb) * MD + col] = f2bf(v);
        }
    }
}

// ---- kernel 4: down GEMM, accumulate all experts in registers -> y (f32) --
// Block tile 128x128, 8 waves (2x4), wave tile 64x32; 16 experts x 44 K-steps.
__global__ __launch_bounds__(256) void moe_down(const bf16* __restrict__ act,
                                                const bf16* __restrict__ wdB,
                                                float* __restrict__ y) {
    __shared__ __align__(16) bf16 As[2][128 * BK];
    __shared__ __align__(16) bf16 Bs[2][128 * BK];

    const int tid  = threadIdx.x;
    const int lane = tid & 31;
    const int wave = tid >> 5;
    const int wm   = wave & 1;
    const int wn   = wave >> 1;
    const int m0   = blockIdx.x * 128;
    const int n0   = blockIdx.y * 128;

    const int row0 = tid >> 2, off = (tid & 3) * 8, row1 = row0 + 64;
    const int NK = MD / BK;            // 44
    const int NS = NEXP * NK;          // 704

#if USE_ASYNC_LDS
    auto issueTile = [&](int ks, int buf) {
        const int e  = ks / NK;
        const int kt = ks - e * NK;
        const bf16* aE = act + (size_t)e * TOK * MD + (size_t)m0 * MD + kt * BK;
        const bf16* bE = wdB + (size_t)e * DIM * MD + (size_t)n0 * MD + kt * BK;
        async_cp16(aE + (size_t)row0 * MD + off, As[buf] + row0 * BK + off);
        async_cp16(aE + (size_t)row1 * MD + off, As[buf] + row1 * BK + off);
        async_cp16(bE + (size_t)row0 * MD + off, Bs[buf] + row0 * BK + off);
        async_cp16(bE + (size_t)row1 * MD + off, Bs[buf] + row1 * BK + off);
    };
#else
    uint4 ra0, ra1, rb0, rb1;
    auto loadRegs = [&](int ks) {
        const int e  = ks / NK;
        const int kt = ks - e * NK;
        const bf16* aE = act + (size_t)e * TOK * MD + (size_t)m0 * MD + kt * BK;
        const bf16* bE = wdB + (size_t)e * DIM * MD + (size_t)n0 * MD + kt * BK;
        ra0 = *(const uint4*)(aE + (size_t)row0 * MD + off);
        ra1 = *(const uint4*)(aE + (size_t)row1 * MD + off);
        rb0 = *(const uint4*)(bE + (size_t)row0 * MD + off);
        rb1 = *(const uint4*)(bE + (size_t)row1 * MD + off);
    };
    auto storeLds = [&](int buf) {
        *(uint4*)(As[buf] + row0 * BK + off) = ra0;
        *(uint4*)(As[buf] + row1 * BK + off) = ra1;
        *(uint4*)(Bs[buf] + row0 * BK + off) = rb0;
        *(uint4*)(Bs[buf] + row1 * BK + off) = rb1;
    };
#endif

    const v8f vz = {0.f,0.f,0.f,0.f,0.f,0.f,0.f,0.f};
    v8f acc[4][2];
#pragma unroll
    for (int mf = 0; mf < 4; ++mf) { acc[mf][0] = vz; acc[mf][1] = vz; }

    int buf = 0;
#if USE_ASYNC_LDS
    issueTile(0, 0);
    for (int ks = 0; ks < NS; ++ks) {
        WAIT_ASYNC0();
        __syncthreads();
        if (ks + 1 < NS) issueTile(ks + 1, buf ^ 1);
#else
    loadRegs(0);
    for (int ks = 0; ks < NS; ++ks) {
        storeLds(buf);
        if (ks + 1 < NS) loadRegs(ks + 1);
        __syncthreads();
#endif
        v16bf fb0 = ldsB(Bs[buf], wn * 32, lane);
        v16bf fb1 = ldsB(Bs[buf], wn * 32 + 16, lane);
#pragma unroll
        for (int mf = 0; mf < 4; ++mf) {
            v16bf fa = ldsA(As[buf], wm * 64 + mf * 16, lane);
            acc[mf][0] = wmma_bf16(fa, fb0, acc[mf][0]);
            acc[mf][1] = wmma_bf16(fa, fb1, acc[mf][1]);
        }
        buf ^= 1;
    }

    const int hi8 = (lane >> 4) * 8, c15 = lane & 15;
#pragma unroll
    for (int mf = 0; mf < 4; ++mf)
#pragma unroll
        for (int nf = 0; nf < 2; ++nf)
#pragma unroll
            for (int r = 0; r < 8; ++r) {
                const int row = m0 + wm * 64 + mf * 16 + hi8 + r;
                const int col = n0 + wn * 32 + nf * 16 + c15;
                y[(size_t)row * DIM + col] = acc[mf][nf][r];
            }
}

// ---------------------------------------------------------------------------
extern "C" void kernel_launch(void* const* d_in, const int* in_sizes, int n_in,
                              void* d_out, int out_size, void* d_ws, size_t ws_size,
                              hipStream_t stream) {
    (void)in_sizes; (void)n_in; (void)out_size; (void)ws_size;
    const float* x      = (const float*)d_in[0];
    const float* gate_w = (const float*)d_in[1];
    const float* w_gate = (const float*)d_in[2];
    const float* w_up   = (const float*)d_in[3];
    const float* w_down = (const float*)d_in[4];
    char* ws = (char*)d_ws;

    bf16* hB  = (bf16*)(ws + H_OFF);
    bf16* wgB = (bf16*)(ws + WG_OFF);
    bf16* wuB = (bf16*)(ws + WU_OFF);
    bf16* wdB = (bf16*)(ws + WD_OFF);
    bf16* actB = (bf16*)(ws + ACT_OFF);
    float* cw  = (float*)(ws + CW_OFF);
    float* y   = (float*)d_out;

    // 1) fp32 -> bf16
    {
        int n4 = (int)(H_ELEMS / 4);
        cvt_f32_bf16<<<(n4 + 255) / 256, 256, 0, stream>>>((const float4*)x, (uint2*)hB, n4);
        n4 = (int)(W_ELEMS / 4);
        cvt_f32_bf16<<<(n4 + 255) / 256, 256, 0, stream>>>((const float4*)w_gate, (uint2*)wgB, n4);
        cvt_f32_bf16<<<(n4 + 255) / 256, 256, 0, stream>>>((const float4*)w_up,   (uint2*)wuB, n4);
        cvt_f32_bf16<<<(n4 + 255) / 256, 256, 0, stream>>>((const float4*)w_down, (uint2*)wdB, n4);
    }
    // 2) router -> dense combine weights
    router_kernel<<<TOK / 8, 256, 0, stream>>>(x, gate_w, cw);

    // 3) gate/up + SwiGLU + cw -> act[e][t][MD] (bf16)
    moe_gateup<<<dim3(TOK / 128, MD / 64, NEXP), 256, 0, stream>>>(
        hB, wgB, wuB, cw, (unsigned short*)actB);

    // 4) down GEMM, sum over experts -> y
    moe_down<<<dim3(TOK / 128, DIM / 128), 256, 0, stream>>>(actB, wdB, y);
}
